// VCLinear_9904194585016
// MI455X (gfx1250) — matrix operand order repaired and checked by reading, not compiled, not run on hardware
//
#include <hip/hip_runtime.h>

typedef __attribute__((ext_vector_type(16))) _Float16 v16h;
typedef __attribute__((ext_vector_type(8)))  _Float16 v8h;
typedef __attribute__((ext_vector_type(8)))  float    v8f;

#define BATCH   32768
#define IN_DIM  1024
#define OUT_DIM 1024
#define XCOLS   1025
#define NB      6

#define BM 256
#define BN 128
#define BK 32
#define ITERS (NB * (IN_DIM / BK))   // 192 flattened (c, k) steps

// ---------------- prep kernels ----------------

// feats (BATCH x 1024, stride 1025 in x) -> f16 row-major stride 1024
__global__ __launch_bounds__(256) void cvt_feats_kernel(const float* __restrict__ x,
                                                        _Float16* __restrict__ a16) {
  size_t idx = (size_t)blockIdx.x * 256 + threadIdx.x;   // < BATCH*IN_DIM
  size_t b = idx >> 10;
  size_t d = idx & 1023;
  a16[idx] = (_Float16)x[b * XCOLS + d];
}

// spline basis per row: [1, t, t^2, relu(t-.25)^2, relu(t-.5)^2, relu(t-.75)^2]
__global__ __launch_bounds__(256) void basis_kernel(const float* __restrict__ x,
                                                    float* __restrict__ basis) {
  int b = blockIdx.x * 256 + threadIdx.x;                 // < BATCH
  float t  = x[(size_t)b * XCOLS + IN_DIM];
  float r1 = fmaxf(t - 0.25f, 0.0f);
  float r2 = fmaxf(t - 0.50f, 0.0f);
  float r3 = fmaxf(t - 0.75f, 0.0f);
  float* o = basis + (size_t)b * NB;
  o[0] = 1.0f;    o[1] = t;       o[2] = t * t;
  o[3] = r1 * r1; o[4] = r2 * r2; o[5] = r3 * r3;
}

// weight [c][k][n] f32 -> f16 transposed [c][n][k]  (so B tiles stage [n][k]-major)
__global__ __launch_bounds__(256) void cvt_w_kernel(const float* __restrict__ w,
                                                    _Float16* __restrict__ wt) {
  size_t idx = (size_t)blockIdx.x * 256 + threadIdx.x;   // < NB*1024*1024
  size_t c = idx >> 20;
  size_t k = (idx >> 10) & 1023;
  size_t n = idx & 1023;
  wt[(c << 20) | (n << 10) | k] = (_Float16)w[idx];
}

// ---------------- WMMA GEMM ----------------
// out[b,e] = sum_c basis[b,c] * (feats @ W_c)[b,e] + bias[e]
// Single GEMM with K = 6*1024; basis folded into A during LDS staging
// (per-row scale), so the compute loop is pure ds_load_b128 + v_wmma.

__global__ __launch_bounds__(256) void vclin_gemm_kernel(
    const _Float16* __restrict__ A16,   // [BATCH][1024]
    const _Float16* __restrict__ WT,    // [6][1024 n][1024 k]
    const float*    __restrict__ basis, // [BATCH][6]
    const float*    __restrict__ bias,  // [1024]
    float*          __restrict__ out)   // [BATCH][1024]
{
  __shared__ __align__(16) _Float16 As[2][BM * BK];  // [m][k], basis-scaled
  __shared__ __align__(16) _Float16 Bs[2][BN * BK];  // [n][k]

  const int tid   = threadIdx.x;
  const int lane  = tid & 31;
  const int wid   = tid >> 5;       // 0..7 (wave32)
  const int wm    = wid >> 1;       // 0..3 : 64-row strip
  const int wn    = wid & 1;        // 0..1 : 64-col strip
  const int laneL = lane & 15;
  const int laneH = lane >> 4;      // 0/1

  const int bm = blockIdx.x * BM;
  const int bn = blockIdx.y * BN;

  // staging coords: chunk = 8 f16; A: 4 chunks/thread, B: 2 chunks/thread
  const int srow = tid >> 2;          // 0..63
  const int sch  = (tid & 3) * 8;     // 0,8,16,24

  v8f acc[4][4];
#pragma unroll
  for (int mi = 0; mi < 4; ++mi)
#pragma unroll
    for (int ni = 0; ni < 4; ++ni)
      acc[mi][ni] = {};

  // stage (c,k)-slab `it` into LDS buffer `buf`; A gets basis[b,c] folded in
  auto stage = [&](int it, int buf) {
    const int c  = it >> 5;
    const int k0 = (it & 31) * BK;
    const _Float16* __restrict__ wc = WT + ((size_t)c << 20);
#pragma unroll
    for (int i = 0; i < 4; ++i) {
      const int row = srow + i * 64;                      // 0..255
      const _Float16 bh = (_Float16)basis[(size_t)(bm + row) * NB + c];
      v8h d = *(const v8h*)&A16[((size_t)(bm + row) << 10) + k0 + sch];
      *(v8h*)&As[buf][row * BK + sch] = d * bh;           // v_pk_mul_f16 x4
    }
#pragma unroll
    for (int i = 0; i < 2; ++i) {
      const int row = srow + i * 64;                      // 0..127
      *(v8h*)&Bs[buf][row * BK + sch] =
          *(const v8h*)&wc[((size_t)(bn + row) << 10) + k0 + sch];
    }
    __builtin_prefetch(&A16[((size_t)(bm + srow) << 10) + k0 + BK + sch], 0, 1);
    __builtin_prefetch(&wc[((size_t)(bn + srow) << 10) + k0 + BK + sch], 0, 1);
  };

  stage(0, 0);
  int p = 0;

  for (int it = 0; it < ITERS; ++it) {
    __syncthreads();                       // staged buffer p is ready
    if (it + 1 < ITERS) stage(it + 1, p ^ 1);   // overlap next slab with WMMA

    const _Float16* __restrict__ as = &As[p][0];
    const _Float16* __restrict__ bs = &Bs[p][0];

    // A fragments: 16x32 f16, M = lane%16 in all VGPRs
    // lanes 0-15 hold K {0..7,16..23}; lanes 16-31 hold K {8..15,24..31}
    v16h af[4];
#pragma unroll
    for (int mi = 0; mi < 4; ++mi) {
      const int ab = (wm * 64 + mi * 16 + laneL) * BK + laneH * 8;
      v8h lo = *(const v8h*)&as[ab];
      v8h hi = *(const v8h*)&as[ab + 16];
      af[mi] = __builtin_shufflevector(lo, hi, 0,1,2,3,4,5,6,7,8,9,10,11,12,13,14,15);
    }

    // B fragments: lanes 0-15 hold K {0..7,16..23}; lanes 16-31 hold K {8..15,24..31}
#pragma unroll
    for (int ni = 0; ni < 4; ++ni) {
      const int bb = (wn * 64 + ni * 16 + laneL) * BK + laneH * 8;
      v8h lo = *(const v8h*)&bs[bb];
      v8h hi = *(const v8h*)&bs[bb + 16];
      v16h bf = __builtin_shufflevector(lo, hi, 0,1,2,3,4,5,6,7,8,9,10,11,12,13,14,15);
#pragma unroll
      for (int mi = 0; mi < 4; ++mi) {
        acc[mi][ni] = __builtin_amdgcn_wmma_f32_16x16x32_f16(
            false, af[mi], false, bf, (short)0, acc[mi][ni], false, false);
      }
    }
    p ^= 1;
  }

  // ---- epilogue: + bias, store f32 ----
  // C/D layout: VGPR i -> row base+i (lanes 0-15) / base+8+i (lanes 16-31), col = lane%16
#pragma unroll
  for (int ni = 0; ni < 4; ++ni) {
    const int col = bn + wn * 64 + ni * 16 + laneL;
    const float bv = bias[col];
#pragma unroll
    for (int mi = 0; mi < 4; ++mi) {
      const int rbase = bm + wm * 64 + mi * 16 + laneH * 8;
#pragma unroll
      for (int i = 0; i < 8; ++i) {
        out[(size_t)(rbase + i) * OUT_DIM + col] = acc[mi][ni][i] + bv;
      }
    }
  }
}

// ---------------- launch ----------------

extern "C" void kernel_launch(void* const* d_in, const int* in_sizes, int n_in,
                              void* d_out, int out_size, void* d_ws, size_t ws_size,
                              hipStream_t stream) {
  const float* x    = (const float*)d_in[0];  // (32768, 1025)
  const float* w    = (const float*)d_in[1];  // (6, 1024, 1024)
  const float* bias = (const float*)d_in[2];  // (1024,)
  float* out = (float*)d_out;                 // (32768, 1024) f32

  // workspace: A16 (64 MiB) | WT f16 transposed (12 MiB) | basis (0.75 MiB)
  _Float16* A16  = (_Float16*)d_ws;
  _Float16* WT   = A16 + (size_t)BATCH * IN_DIM;
  float*    bas  = (float*)(WT + (size_t)NB * IN_DIM * OUT_DIM);

  cvt_feats_kernel<<<(BATCH * IN_DIM) / 256, 256, 0, stream>>>(x, A16);
  basis_kernel<<<BATCH / 256, 256, 0, stream>>>(x, bas);
  cvt_w_kernel<<<(NB * IN_DIM * OUT_DIM) / 256, 256, 0, stream>>>(w, WT);

  dim3 grid(BATCH / BM, OUT_DIM / BN);   // 128 x 8 blocks
  vclin_gemm_kernel<<<grid, 256, 0, stream>>>(A16, WT, bas, bias, out);
}
